// EncoderBlock_18124761989659
// MI455X (gfx1250) — matrix operand order repaired and checked by reading, not compiled, not run on hardware
//
#include <hip/hip_runtime.h>

// ---------------------------------------------------------------------------
// MI455X (gfx1250) encoder block: all GEMMs via v_wmma_f32_16x16x32_bf16,
// f32 accumulate; softmax / layernorm / bias / residual in fp32.
// Softmax row-max via DPP (VALU, co-executes with WMMA); softmax row-sum via
// an extra WMMA against an all-ones B fragment (no cross-lane shuffles).
// ---------------------------------------------------------------------------

typedef __bf16 bf16;
typedef __attribute__((ext_vector_type(16))) __bf16 v16bf;
typedef __attribute__((ext_vector_type(8)))  float  v8f;

#define TOK   4096      // N*SEQ
#define MDIM  768
#define FFD   3072
#define HEADS 12
#define NBAT  4
#define SEQ   1024
#define DHEAD 64

static __device__ __forceinline__ v8f wmma_bf16(v16bf a, v16bf b, v8f c) {
  return __builtin_amdgcn_wmma_f32_16x16x32_bf16(false, a, false, b, (short)0, c,
                                                 false, false);
}

// max across each 16-lane row (C/D layout halves) using DPP only
static __device__ __forceinline__ float dpp_rowmax16(float x) {
  int t;
  t = __builtin_amdgcn_update_dpp(__float_as_int(x), __float_as_int(x),
                                  0x0B1 /*quad_perm(1,0,3,2)*/, 0xf, 0xf, false);
  x = fmaxf(x, __int_as_float(t));
  t = __builtin_amdgcn_update_dpp(__float_as_int(x), __float_as_int(x),
                                  0x04E /*quad_perm(2,3,0,1)*/, 0xf, 0xf, false);
  x = fmaxf(x, __int_as_float(t));
  t = __builtin_amdgcn_update_dpp(__float_as_int(x), __float_as_int(x),
                                  0x141 /*row_half_mirror*/, 0xf, 0xf, false);
  x = fmaxf(x, __int_as_float(t));
  t = __builtin_amdgcn_update_dpp(__float_as_int(x), __float_as_int(x),
                                  0x140 /*row_mirror*/, 0xf, 0xf, false);
  x = fmaxf(x, __int_as_float(t));
  return x;
}

// ---------------------------------------------------------------------------
// Conversion / permutation helpers
// ---------------------------------------------------------------------------
__global__ void cvt_f32_bf16(const float* __restrict__ in, bf16* __restrict__ out, int n) {
  int i = blockIdx.x * 256 + threadIdx.x;
  if (i < n) out[i] = (bf16)in[i];
}

// (H, M, D) f32 -> (M, H*D) bf16
__global__ void permute_qkv(const float* __restrict__ in, bf16* __restrict__ out) {
  int i = blockIdx.x * 256 + threadIdx.x;
  if (i < MDIM * MDIM) {
    int m = i / MDIM, c = i % MDIM;
    int h = c >> 6, d = c & 63;
    out[i] = (bf16)in[((size_t)h * MDIM + m) * DHEAD + d];
  }
}

// ---------------------------------------------------------------------------
// Generic bf16 WMMA GEMM: C(4096 x Ntot) = A(4096 x Kdim) * B(Kdim x Ntot)
// block = 128 threads (4 waves); block tile 128x64; wave tile 32x64; K step 32.
// 8 WMMAs per wave per K-step against 6 LDS fragment loads.
// MODE 0: QKV scatter (+bias) -> bf16 [h*N+n][1024][64]
// MODE 1/3: +bias +residual(f32) -> f32 row-major 768 (pre-layernorm sum)
// MODE 2: +bias, ReLU -> bf16 row-major Ntot
// ---------------------------------------------------------------------------
template <int MODE>
__global__ __launch_bounds__(128) void gemm_bf16(
    const bf16* __restrict__ A, const bf16* __restrict__ B,
    const float* __restrict__ bias, const float* __restrict__ resid,
    void* __restrict__ out, int Kdim, int Ntot) {
  __shared__ __align__(16) bf16 As[128][32];
  __shared__ __align__(16) bf16 Bs[32][64];

  const int tid = threadIdx.x;
  const int lane = tid & 31, w = tid >> 5, hf = lane >> 4;
  const int r0 = blockIdx.x * 128;
  const int n0 = blockIdx.y * 64;

  v8f acc[2][4];
#pragma unroll
  for (int r = 0; r < 2; ++r)
#pragma unroll
    for (int t = 0; t < 4; ++t) acc[r][t] = v8f{};

  for (int k0 = 0; k0 < Kdim; k0 += 32) {
    if (k0 + 32 < Kdim)  // speculative prefetch of next A tile
      __builtin_prefetch(&A[(size_t)(r0 + tid) * Kdim + k0 + 32], 0, 0);
    // stage A (128x32) and B (32x64) tiles as 16B chunks
    for (int c = tid; c < 512; c += 128) {
      int ar = c >> 2, as = c & 3;
      *(uint4*)&As[ar][as * 8] =
          *(const uint4*)&A[(size_t)(r0 + ar) * Kdim + k0 + as * 8];
    }
    for (int c = tid; c < 256; c += 128) {
      int br = c >> 3, bs = c & 7;
      *(uint4*)&Bs[br][bs * 8] =
          *(const uint4*)&B[(size_t)(k0 + br) * Ntot + n0 + bs * 8];
    }
    __syncthreads();

    // two A fragments (16x32 bf16 each) for this wave's 32 rows
    v16bf a[2];
#pragma unroll
    for (int r = 0; r < 2; ++r) {
      const int mrow = (w << 5) + (r << 4) + (lane & 15);
#pragma unroll
      for (int j = 0; j < 8; ++j) {
        int kk = ((j >> 2) << 4) + (hf << 3) + ((j & 3) << 1);
        a[r][2 * j] = As[mrow][kk];
        a[r][2 * j + 1] = As[mrow][kk + 1];
      }
    }
    // 4 B fragments (32x16 each), each reused by both A fragments
#pragma unroll
    for (int t = 0; t < 4; ++t) {
      v16bf b;
      int n = (t << 4) + (lane & 15);
#pragma unroll
      for (int j = 0; j < 8; ++j) {
        int kk = ((j >> 2) << 4) + (hf << 3) + ((j & 3) << 1);
        b[2 * j] = Bs[kk][n];
        b[2 * j + 1] = Bs[kk + 1][n];
      }
#pragma unroll
      for (int r = 0; r < 2; ++r) acc[r][t] = wmma_bf16(a[r], b, acc[r][t]);
    }
    __syncthreads();
  }

  // epilogue: C layout = VGPR e -> M = e + hf*8, N = lane&15 (+16*t)
#pragma unroll
  for (int r = 0; r < 2; ++r) {
    const int rbase = r0 + (w << 5) + (r << 4) + (hf << 3);
#pragma unroll
    for (int t = 0; t < 4; ++t) {
      const int col = n0 + t * 16 + (lane & 15);
      const float bval = bias[col];
#pragma unroll
      for (int e = 0; e < 8; ++e) {
        const int row = rbase + e;
        float v = acc[r][t][e] + bval;
        if (MODE == 0) {
          int h = col >> 6, d = col & 63;
          int n = row >> 10, kk = row & 1023;
          ((bf16*)out)[(((size_t)(h * NBAT + n) * SEQ + kk) << 6) + d] = (bf16)v;
        } else if (MODE == 2) {
          ((bf16*)out)[(size_t)row * Ntot + col] = (bf16)(v > 0.f ? v : 0.f);
        } else {  // MODE 1 / 3: fp32 pre-layernorm sum (GEMM + bias + residual)
          ((float*)out)[(size_t)row * MDIM + col] =
              v + resid[(size_t)row * MDIM + col];
        }
      }
    }
  }
}

// ---------------------------------------------------------------------------
// Flash attention: block = 128 threads (4 waves), 64 query rows of one (h,n).
// K/V staged in LDS per 32-key chunk; online softmax in fp32.
// Row max via DPP; row sum via WMMA P @ ones (5th accumulator tile).
// q/k/v layout: [h*N+n][1024][64] bf16.  y: [4096][768] bf16 (heads concat).
// ---------------------------------------------------------------------------
__global__ __launch_bounds__(128) void flash_attn(
    const bf16* __restrict__ q, const bf16* __restrict__ k,
    const bf16* __restrict__ v, bf16* __restrict__ y) {
  __shared__ __align__(16) bf16 Kt[32][64];
  __shared__ __align__(16) bf16 Vt[32][64];
  __shared__ __align__(16) bf16 Ps[4][16][32];

  const int tid = threadIdx.x;
  const int lane = tid & 31, w = tid >> 5, hf = lane >> 4;
  const int hn = blockIdx.y;   // h*NBAT + n
  const int rb = blockIdx.x;   // 64-row block within sequence
  const size_t base = (size_t)hn * SEQ * DHEAD;

  // per-wave Q fragments (16 rows x 64 dims -> two 16x32 A-frags), loaded once,
  // pre-scaled by 1/sqrt(D) so the S tiles need no per-chunk scaling
  const int qrow = rb * 64 + w * 16 + (lane & 15);
  const bf16* qp = q + base + (size_t)qrow * DHEAD;
  v16bf qa0, qa1;
#pragma unroll
  for (int j = 0; j < 8; ++j) {
    int kk = ((j >> 2) << 4) + (hf << 3) + ((j & 3) << 1);
    qa0[2 * j]     = (bf16)(0.125f * (float)qp[kk]);
    qa0[2 * j + 1] = (bf16)(0.125f * (float)qp[kk + 1]);
    qa1[2 * j]     = (bf16)(0.125f * (float)qp[32 + kk]);
    qa1[2 * j + 1] = (bf16)(0.125f * (float)qp[32 + kk + 1]);
  }
  v16bf onesf;
#pragma unroll
  for (int i = 0; i < 16; ++i) onesf[i] = (bf16)1.0f;

  v8f o0 = {}, o1 = {}, o2 = {}, o3 = {}, ol = {};
  float mrow[8];
#pragma unroll
  for (int e = 0; e < 8; ++e) mrow[e] = -1e30f;

  for (int kb = 0; kb < SEQ / 32; ++kb) {
    const bf16* kg = k + base + (size_t)kb * 32 * DHEAD;
    const bf16* vg = v + base + (size_t)kb * 32 * DHEAD;
    for (int c = tid; c < 256; c += 128) {
      int row = c >> 3, seg = c & 7;
      *(uint4*)&Kt[row][seg * 8] = *(const uint4*)&kg[row * 64 + seg * 8];
      *(uint4*)&Vt[row][seg * 8] = *(const uint4*)&vg[row * 64 + seg * 8];
    }
    __syncthreads();

    // S = Q (16x64) * K^T (64x32): B[kk][n] = Kt[key][d=kk]
    v8f s0 = {}, s1 = {};
    {
      const int n = lane & 15;
      v16bf b00, b01, b10, b11;
#pragma unroll
      for (int j = 0; j < 8; ++j) {
        int kk = ((j >> 2) << 4) + (hf << 3) + ((j & 3) << 1);
        b00[2 * j] = Kt[n][kk];           b00[2 * j + 1] = Kt[n][kk + 1];
        b01[2 * j] = Kt[n][32 + kk];      b01[2 * j + 1] = Kt[n][32 + kk + 1];
        b10[2 * j] = Kt[16 + n][kk];      b10[2 * j + 1] = Kt[16 + n][kk + 1];
        b11[2 * j] = Kt[16 + n][32 + kk]; b11[2 * j + 1] = Kt[16 + n][32 + kk + 1];
      }
      s0 = wmma_bf16(qa0, b00, s0);
      s0 = wmma_bf16(qa1, b01, s0);
      s1 = wmma_bf16(qa0, b10, s1);
      s1 = wmma_bf16(qa1, b11, s1);
    }

    // online softmax (per C-component row: M = e + hf*8; cols span 16 lanes)
#pragma unroll
    for (int e = 0; e < 8; ++e) {
      float a0 = s0[e], a1 = s1[e];
      float mx = dpp_rowmax16(fmaxf(a0, a1));
      float mn = fmaxf(mrow[e], mx);
      float alpha = __expf(mrow[e] - mn);
      float p0 = __expf(a0 - mn), p1 = __expf(a1 - mn);
      mrow[e] = mn;
      o0[e] *= alpha; o1[e] *= alpha; o2[e] *= alpha; o3[e] *= alpha;
      ol[e] *= alpha;  // softmax denominator tile rescales identically
      const int prow = e + hf * 8;  // C-layout row -> P buffer row
      Ps[w][prow][lane & 15] = (bf16)p0;
      Ps[w][prow][16 + (lane & 15)] = (bf16)p1;
    }
    // per-wave LDS RAW: drain our DS stores before re-reading Ps[w]
    asm volatile("s_wait_dscnt 0" ::: "memory");

    // O += P (16x32) * V (32x64);  L += P * ones (row-sum of P, every column)
    v16bf pa;
#pragma unroll
    for (int j = 0; j < 8; ++j) {
      int kk = ((j >> 2) << 4) + (hf << 3) + ((j & 3) << 1);
      pa[2 * j] = Ps[w][lane & 15][kk];
      pa[2 * j + 1] = Ps[w][lane & 15][kk + 1];
    }
    ol = wmma_bf16(pa, onesf, ol);
#pragma unroll
    for (int t = 0; t < 4; ++t) {
      v16bf vb;
      const int n = (t << 4) + (lane & 15);
#pragma unroll
      for (int j = 0; j < 8; ++j) {
        int kk = ((j >> 2) << 4) + (hf << 3) + ((j & 3) << 1);
        vb[2 * j] = Vt[kk][n];
        vb[2 * j + 1] = Vt[kk + 1][n];
      }
      if (t == 0) o0 = wmma_bf16(pa, vb, o0);
      if (t == 1) o1 = wmma_bf16(pa, vb, o1);
      if (t == 2) o2 = wmma_bf16(pa, vb, o2);
      if (t == 3) o3 = wmma_bf16(pa, vb, o3);
    }
    __syncthreads();  // protect Kt/Vt before next chunk's staging
  }

  // normalize and store concat-head output (bf16, row-major 768)
  const int hh = hn >> 2, nn = hn & 3;
  const int rbase = rb * 64 + w * 16 + (hf << 3);
#pragma unroll
  for (int e = 0; e < 8; ++e) {
    const float inv = 1.0f / ol[e];
    const int grow = nn * SEQ + rbase + e;
    const size_t o = (size_t)grow * MDIM + hh * DHEAD + (lane & 15);
    y[o] = (bf16)(o0[e] * inv);
    y[o + 16] = (bf16)(o1[e] * inv);
    y[o + 32] = (bf16)(o2[e] * inv);
    y[o + 48] = (bf16)(o3[e] * inv);
  }
}

// ---------------------------------------------------------------------------
// LayerNorm over 768 features; one block (256 threads) per row.
// ---------------------------------------------------------------------------
__global__ __launch_bounds__(256) void layernorm_768(
    const float* __restrict__ in, const float* __restrict__ g,
    const float* __restrict__ be, float* __restrict__ outf,
    bf16* __restrict__ outb) {
  __shared__ float red[8];
  __shared__ float stat[2];
  const int row = blockIdx.x, tid = threadIdx.x;
  const float* x = in + (size_t)row * MDIM;

  float s = 0.f;
  for (int c = tid; c < MDIM; c += 256) s += x[c];
#pragma unroll
  for (int d = 16; d >= 1; d >>= 1) s += __shfl_xor(s, d, 32);
  if ((tid & 31) == 0) red[tid >> 5] = s;
  __syncthreads();
  if (tid == 0) {
    float t = 0.f;
    for (int i = 0; i < 8; ++i) t += red[i];
    stat[0] = t * (1.0f / MDIM);
  }
  __syncthreads();
  const float mean = stat[0];

  float vs = 0.f;
  for (int c = tid; c < MDIM; c += 256) {
    float d = x[c] - mean;
    vs += d * d;
  }
#pragma unroll
  for (int d = 16; d >= 1; d >>= 1) vs += __shfl_xor(vs, d, 32);
  __syncthreads();
  if ((tid & 31) == 0) red[tid >> 5] = vs;
  __syncthreads();
  if (tid == 0) {
    float t = 0.f;
    for (int i = 0; i < 8; ++i) t += red[i];
    stat[1] = rsqrtf(t * (1.0f / MDIM) + 1e-10f);
  }
  __syncthreads();
  const float rstd = stat[1];

  for (int c = tid; c < MDIM; c += 256) {
    float vv = (x[c] - mean) * rstd * g[c] + be[c];
    outf[(size_t)row * MDIM + c] = vv;
    if (outb) outb[(size_t)row * MDIM + c] = (bf16)vv;
  }
}

// ---------------------------------------------------------------------------
// Orchestration
// ---------------------------------------------------------------------------
extern "C" void kernel_launch(void* const* d_in, const int* in_sizes, int n_in,
                              void* d_out, int out_size, void* d_ws,
                              size_t ws_size, hipStream_t stream) {
  (void)in_sizes; (void)n_in; (void)out_size; (void)ws_size;
  const float* x   = (const float*)d_in[0];
  const float* Wq  = (const float*)d_in[1];
  const float* bq  = (const float*)d_in[2];
  const float* Wk  = (const float*)d_in[3];
  const float* bk  = (const float*)d_in[4];
  const float* Wv  = (const float*)d_in[5];
  const float* bv  = (const float*)d_in[6];
  const float* Wo  = (const float*)d_in[7];
  const float* bo  = (const float*)d_in[8];
  const float* g1  = (const float*)d_in[9];
  const float* be1 = (const float*)d_in[10];
  const float* g2  = (const float*)d_in[11];
  const float* be2 = (const float*)d_in[12];
  const float* W1  = (const float*)d_in[13];
  const float* bf1 = (const float*)d_in[14];
  const float* W2  = (const float*)d_in[15];
  const float* bf2 = (const float*)d_in[16];
  float* out = (float*)d_out;

  char* ws = (char*)d_ws;
  size_t off = 0;
  auto carve = [&](size_t bytes) -> void* {
    void* p = ws + off;
    off = (off + bytes + 255) & ~(size_t)255;
    return p;
  };
  bf16* xb    = (bf16*)carve((size_t)TOK * MDIM * 2);
  bf16* wq_b  = (bf16*)carve((size_t)MDIM * MDIM * 2);
  bf16* wk_b  = (bf16*)carve((size_t)MDIM * MDIM * 2);
  bf16* wv_b  = (bf16*)carve((size_t)MDIM * MDIM * 2);
  bf16* wo_b  = (bf16*)carve((size_t)MDIM * MDIM * 2);
  bf16* w1_b  = (bf16*)carve((size_t)MDIM * FFD * 2);
  bf16* w2_b  = (bf16*)carve((size_t)FFD * MDIM * 2);
  bf16* qb    = (bf16*)carve((size_t)TOK * MDIM * 2);
  bf16* kb    = (bf16*)carve((size_t)TOK * MDIM * 2);
  bf16* vb    = (bf16*)carve((size_t)TOK * MDIM * 2);
  bf16* ybf   = (bf16*)carve((size_t)TOK * MDIM * 2);
  float* sum1 = (float*)carve((size_t)TOK * MDIM * 4);
  float* out1f= (float*)carve((size_t)TOK * MDIM * 4);
  bf16* out1b = (bf16*)carve((size_t)TOK * MDIM * 2);
  bf16* ffb   = (bf16*)carve((size_t)TOK * FFD * 2);
  float* sum2 = (float*)carve((size_t)TOK * MDIM * 4);

  // 1) precision conversion / weight permutation
  cvt_f32_bf16<<<(TOK * MDIM + 255) / 256, 256, 0, stream>>>(x, xb, TOK * MDIM);
  permute_qkv<<<(MDIM * MDIM + 255) / 256, 256, 0, stream>>>(Wq, wq_b);
  permute_qkv<<<(MDIM * MDIM + 255) / 256, 256, 0, stream>>>(Wk, wk_b);
  permute_qkv<<<(MDIM * MDIM + 255) / 256, 256, 0, stream>>>(Wv, wv_b);
  cvt_f32_bf16<<<(MDIM * MDIM + 255) / 256, 256, 0, stream>>>(Wo, wo_b, MDIM * MDIM);
  cvt_f32_bf16<<<(MDIM * FFD + 255) / 256, 256, 0, stream>>>(W1, w1_b, MDIM * FFD);
  cvt_f32_bf16<<<(FFD * MDIM + 255) / 256, 256, 0, stream>>>(W2, w2_b, FFD * MDIM);

  // 2) Q/K/V projections (scatter into [h*N+n][1024][64] bf16)
  dim3 gP(TOK / 128, MDIM / 64);
  gemm_bf16<0><<<gP, 128, 0, stream>>>(xb, wq_b, bq, nullptr, qb, MDIM, MDIM);
  gemm_bf16<0><<<gP, 128, 0, stream>>>(xb, wk_b, bk, nullptr, kb, MDIM, MDIM);
  gemm_bf16<0><<<gP, 128, 0, stream>>>(xb, wv_b, bv, nullptr, vb, MDIM, MDIM);

  // 3) flash attention -> ybf (concat heads, row-major 768)
  flash_attn<<<dim3(SEQ / 64, HEADS * NBAT), 128, 0, stream>>>(qb, kb, vb, ybf);

  // 4) output projection + bias + residual(x) -> sum1 (f32), then LN1
  gemm_bf16<1><<<gP, 128, 0, stream>>>(ybf, wo_b, bo, x, sum1, MDIM, MDIM);
  layernorm_768<<<TOK, 256, 0, stream>>>(sum1, g1, be1, out1f, out1b);

  // 5) FFN: relu(out1 @ W1 + bf1) @ W2 + bf2, residual(out1), LN2 -> out
  gemm_bf16<2><<<dim3(TOK / 128, FFD / 64), 128, 0, stream>>>(
      out1b, w1_b, bf1, nullptr, ffb, MDIM, FFD);
  gemm_bf16<3><<<gP, 128, 0, stream>>>(ffb, w2_b, bf2, out1f, sum2, FFD, MDIM);
  layernorm_768<<<TOK, 256, 0, stream>>>(sum2, g2, be2, out, nullptr);
}
